// UNet_41858751266869
// MI455X (gfx1250) — matrix operand order repaired and checked by reading, not compiled, not run on hardware
//
#include <hip/hip_runtime.h>

typedef __attribute__((ext_vector_type(16))) __bf16 v16bf;
typedef __attribute__((ext_vector_type(8)))  __bf16 v8bf;
typedef __attribute__((ext_vector_type(8)))  float  v8f;

#define EPSF 1e-4f

// ---------------------------------------------------------------------------
// Deterministic BN reduction: per-block partial sum / sumsq per channel.
// Supports a two-source "concat" view: channels [0,CA) from a, [CA,C) from b.
// partial layout: [gridDim][128] sums followed by [gridDim][128] sumsqs.
// ---------------------------------------------------------------------------
__global__ void bn_reduce_kernel(const float* __restrict__ a, const float* __restrict__ b,
                                 int CA, int C, int R, float* __restrict__ partial)
{
  __shared__ float shs[256];
  __shared__ float shq[256];
  int t = threadIdx.x;
  int c = t & (C - 1);
  int rpb = blockDim.x / C;
  int j = t / C;
  int CB = C - CA;
  float s = 0.f, q = 0.f;
  for (int r = blockIdx.x * rpb + j; r < R; r += gridDim.x * rpb) {
    float x = (c < CA) ? a[(size_t)r * CA + c] : b[(size_t)r * CB + (c - CA)];
    s += x; q += x * x;
  }
  shs[t] = s; shq[t] = q;
  __syncthreads();
  if (t < C) {
    float ts = 0.f, tq = 0.f;
    for (int jj = 0; jj < rpb; ++jj) { ts += shs[t + jj * C]; tq += shq[t + jj * C]; }
    partial[(size_t)blockIdx.x * 128 + t] = ts;
    partial[(size_t)(gridDim.x + blockIdx.x) * 128 + t] = tq;
  }
}

// ss[c] = g*rsqrt(var+eps), ss[64+c] = b - mu*scale
__global__ void bn_finalize_kernel(const float* __restrict__ partial, int NB,
                                   const float* __restrict__ g, const float* __restrict__ bia,
                                   int C, float invR, float* __restrict__ ss)
{
  int c = threadIdx.x;
  if (c >= C) return;
  float s = 0.f, q = 0.f;
  for (int nb = 0; nb < NB; ++nb) {
    s += partial[(size_t)nb * 128 + c];
    q += partial[(size_t)(NB + nb) * 128 + c];
  }
  float mu  = s * invR;
  float var = q * invR - mu * mu;
  float sc  = g[c] * rsqrtf(var + EPSF);
  ss[c]      = sc;
  ss[64 + c] = bia[c] - mu * sc;
}

// Fused BN+ReLU + convert-to-bf16 into padded feature buffer [(P+1) x C];
// row P is zeroed (sentinel row). Two-source concat view like bn_reduce.
__global__ void bn_relu_cvt_kernel(const float* __restrict__ a, const float* __restrict__ b,
                                   int CA, int C, int P, const float* __restrict__ ss,
                                   __bf16* __restrict__ out)
{
  size_t total = (size_t)(P + 1) * C;
  int CB = C - CA;
  for (size_t i = (size_t)blockIdx.x * blockDim.x + threadIdx.x; i < total;
       i += (size_t)gridDim.x * blockDim.x) {
    int r = (int)(i / (size_t)C);
    int c = (int)(i - (size_t)r * C);
    float y = 0.f;
    if (r < P) {
      float x = (c < CA) ? a[(size_t)r * CA + c] : b[(size_t)r * CB + (c - CA)];
      y = fmaf(x, ss[c], ss[64 + c]);
      y = y > 0.f ? y : 0.f;
    }
    out[i] = (__bf16)y;
  }
}

// Build 8-tap gather indices for the transpose conv:
// didx[k][n] = (up_k[n]==k) ? up_cidx[n] : M (sentinel -> zero row)
__global__ void build_deconv_idx_kernel(const int* __restrict__ up_cidx,
                                        const int* __restrict__ up_k,
                                        int N, int M, int* __restrict__ didx)
{
  int n = blockIdx.x * blockDim.x + threadIdx.x;
  if (n >= N) return;
  int ci = up_cidx[n], kk = up_k[n];
  for (int k = 0; k < 8; ++k) didx[(size_t)k * N + n] = (kk == k) ? ci : M;
}

// Pack fp32 weights [taps][Cin][Cout] into bf16 WMMA B-fragment order:
// dst[((k*KC+kc)*CT+ct)*32 + lane][e] = W[k][kc*32 + (e>>3)*16 + (lane>>4)*8 + (e&7)][ct*16 + (lane&15)]
__global__ void pack_w_kernel(const float* __restrict__ w, int taps, int Cin, int Cout,
                              __bf16* __restrict__ dst)
{
  int KC = Cin >> 5, CT = Cout >> 4;
  size_t total = (size_t)taps * Cin * Cout;
  for (size_t i = (size_t)blockIdx.x * blockDim.x + threadIdx.x; i < total;
       i += (size_t)gridDim.x * blockDim.x) {
    int e    = (int)(i & 15);
    int lane = (int)((i >> 4) & 31);
    size_t rest = i >> 9;
    int ct = (int)(rest % CT);
    size_t rest2 = rest / CT;
    int kc = (int)(rest2 % KC);
    int k  = (int)(rest2 / KC);
    int K   = kc * 32 + (e >> 3) * 16 + (lane >> 4) * 8 + (e & 7);
    int col = ct * 16 + (lane & 15);
    dst[i] = (__bf16)w[((size_t)k * Cin + K) * Cout + col];
  }
}

// ---------------------------------------------------------------------------
// Generic gather-GEMM sparse conv: out[Mrows,COUT] = sum_k feat[nbr[k]] @ Wk
// One wave handles T consecutive 16-row tiles (B fragments reused T times).
// bf16 WMMA 16x16x32, fp32 accumulate.
// feat has P+1 rows, row P = zeros (sentinel index == P).
// ---------------------------------------------------------------------------
template<int CIN, int COUT, int T>
__global__ __launch_bounds__(128) void conv_wmma_kernel(
    const __bf16* __restrict__ feat, const int* __restrict__ nbr,
    const __bf16* __restrict__ wp, float* __restrict__ out,
    int Mrows, int P, int taps)
{
  constexpr int KC = CIN / 32;
  constexpr int CT = COUT / 16;
  const int lane  = threadIdx.x & 31;
  const int wave  = threadIdx.x >> 5;
  const int group = blockIdx.x * 4 + wave;
  const int base  = group * (16 * T);
  if (base >= Mrows) return;          // uniform across the wave
  const int r = lane & 15;
  const int h = lane >> 4;

  v8f acc[T][CT] = {};

  for (int k = 0; k < taps; ++k) {
    const int* nk = nbr + (size_t)k * Mrows;
    int idx[T];
    #pragma unroll
    for (int t = 0; t < T; ++t) {
      int row = base + t * 16 + r;
      idx[t] = (row < Mrows) ? nk[row] : P;
    }
    const __bf16* wk = wp + (size_t)k * (CIN * COUT);
    #pragma unroll
    for (int kc = 0; kc < KC; ++kc) {
      // A-fragment (16-bit A 16x32 layout): lane (r,h) takes
      // elements 0..7  -> K = 32*kc + 8*h + i
      // elements 8..15 -> K = 32*kc + 16 + 8*h + (i-8)
      v16bf af[T];
      #pragma unroll
      for (int t = 0; t < T; ++t) {
        const __bf16* ap = feat + (size_t)idx[t] * CIN + kc * 32 + h * 8;
        v8bf lo = *(const v8bf*)(ap);
        v8bf hi = *(const v8bf*)(ap + 16);
        af[t] = __builtin_shufflevector(lo, hi,
            0,1,2,3,4,5,6,7,8,9,10,11,12,13,14,15);
      }
      #pragma unroll
      for (int ct = 0; ct < CT; ++ct) {
        v16bf bfrag = *(const v16bf*)(wk + (size_t)((kc * CT + ct) * 32 + lane) * 16);
        #pragma unroll
        for (int t = 0; t < T; ++t) {
          acc[t][ct] = __builtin_amdgcn_wmma_f32_16x16x32_bf16(
              false, af[t], false, bfrag, (short)0, acc[t][ct], false, false);
        }
      }
    }
  }

  // D layout: element e of lane (r,h) -> row tilebase+e+8h, col ct*16+r
  #pragma unroll
  for (int t = 0; t < T; ++t) {
    #pragma unroll
    for (int ct = 0; ct < CT; ++ct) {
      #pragma unroll
      for (int e = 0; e < 8; ++e) {
        int orow = base + t * 16 + e + 8 * h;
        if (orow < Mrows) out[(size_t)orow * COUT + ct * 16 + r] = acc[t][ct][e];
      }
    }
  }
}

template<int CIN, int COUT>
static inline void launch_conv(const __bf16* feat, const int* nbr, const __bf16* wp,
                               float* out, int Mrows, int P, int taps, hipStream_t stream)
{
  constexpr int T = 4;
  int groups = (Mrows + 16 * T - 1) / (16 * T);
  int blocks = (groups + 3) / 4;
  conv_wmma_kernel<CIN, COUT, T><<<blocks, 128, 0, stream>>>(feat, nbr, wp, out, Mrows, P, taps);
}

// ---------------------------------------------------------------------------
extern "C" void kernel_launch(void* const* d_in, const int* in_sizes, int n_in,
                              void* d_out, int out_size, void* d_ws, size_t ws_size,
                              hipStream_t stream)
{
  (void)n_in; (void)out_size; (void)ws_size;

  const float* feat   = (const float*)d_in[0];
  const float* w_sub1 = (const float*)d_in[1];
  const float* w_down = (const float*)d_in[2];
  const float* w_sub2 = (const float*)d_in[3];
  const float* w_up   = (const float*)d_in[4];
  const float* w_sub3 = (const float*)d_in[5];
  const float* g1 = (const float*)d_in[6],  *b1 = (const float*)d_in[7];
  const float* g2 = (const float*)d_in[8],  *b2 = (const float*)d_in[9];
  const float* g3 = (const float*)d_in[10], *b3 = (const float*)d_in[11];
  const float* g4 = (const float*)d_in[12], *b4 = (const float*)d_in[13];
  const float* g5 = (const float*)d_in[14], *b5 = (const float*)d_in[15];
  const int* nbr_fine   = (const int*)d_in[16];
  const int* nbr_coarse = (const int*)d_in[17];
  const int* down_idx   = (const int*)d_in[18];
  const int* up_cidx    = (const int*)d_in[19];
  const int* up_k       = (const int*)d_in[20];

  const int N  = in_sizes[0] / 32;     // fine voxels
  const int M  = in_sizes[18] / 8;     // coarse voxels
  const int PM = (N > M) ? N : M;

  // ---- workspace carving (256B aligned) ----
  char* ws = (char*)d_ws;
  size_t off = 0;
  auto take = [&](size_t bytes) -> char* {
    char* p = ws + off;
    off = (off + bytes + 255) & ~(size_t)255;
    return p;
  };
  float*  ss      = (float*)take((size_t)5 * 128 * sizeof(float));
  float*  partial = (float*)take((size_t)2 * 128 * 128 * sizeof(float));
  __bf16* wp1  = (__bf16*)take((size_t)27 * 32 * 32 * 2);
  __bf16* wpd  = (__bf16*)take((size_t)8  * 32 * 64 * 2);
  __bf16* wp2  = (__bf16*)take((size_t)27 * 64 * 64 * 2);
  __bf16* wpu  = (__bf16*)take((size_t)8  * 64 * 32 * 2);
  __bf16* wp3  = (__bf16*)take((size_t)27 * 64 * 32 * 2);
  int*    didx = (int*)  take((size_t)8 * N * 4);
  float*  x1   = (float*)take((size_t)N * 32 * 4);
  float*  xc   = (float*)take((size_t)M * 64 * 4);
  float*  xf   = (float*)take((size_t)N * 32 * 4);
  __bf16* fbuf = (__bf16*)take((size_t)(PM + 1) * 64 * 2);

  auto cvt_blocks = [](size_t total) -> int {
    size_t b = (total + 255) / 256;
    return (int)(b > 4096 ? 4096 : b);
  };

  // ---- one-time-per-call prep ----
  pack_w_kernel<<<128, 256, 0, stream>>>(w_sub1, 27, 32, 32, wp1);
  pack_w_kernel<<<128, 256, 0, stream>>>(w_down,  8, 32, 64, wpd);
  pack_w_kernel<<<256, 256, 0, stream>>>(w_sub2, 27, 64, 64, wp2);
  pack_w_kernel<<<128, 256, 0, stream>>>(w_up,    8, 64, 32, wpu);
  pack_w_kernel<<<256, 256, 0, stream>>>(w_sub3, 27, 64, 32, wp3);
  build_deconv_idx_kernel<<<(N + 255) / 256, 256, 0, stream>>>(up_cidx, up_k, N, M, didx);

  // ---- stage 1: BN+ReLU(feat) -> 27-tap sub conv 32->32 -> x1 (skip) ----
  bn_reduce_kernel<<<128, 256, 0, stream>>>(feat, feat, 32, 32, N, partial);
  bn_finalize_kernel<<<1, 64, 0, stream>>>(partial, 128, g1, b1, 32, 1.0f / N, ss + 0 * 128);
  bn_relu_cvt_kernel<<<cvt_blocks((size_t)(N + 1) * 32), 256, 0, stream>>>(
      feat, feat, 32, 32, N, ss + 0 * 128, fbuf);
  launch_conv<32, 32>(fbuf, nbr_fine, wp1, x1, N, N, 27, stream);

  // ---- stage 2: BN+ReLU(x1) -> 8-tap down conv 32->64 -> xc ----
  bn_reduce_kernel<<<128, 256, 0, stream>>>(x1, x1, 32, 32, N, partial);
  bn_finalize_kernel<<<1, 64, 0, stream>>>(partial, 128, g2, b2, 32, 1.0f / N, ss + 1 * 128);
  bn_relu_cvt_kernel<<<cvt_blocks((size_t)(N + 1) * 32), 256, 0, stream>>>(
      x1, x1, 32, 32, N, ss + 1 * 128, fbuf);
  launch_conv<32, 64>(fbuf, down_idx, wpd, xc, M, N, 8, stream);

  // ---- stage 3: BN+ReLU(xc) -> 27-tap sub conv 64->64 -> xc ----
  bn_reduce_kernel<<<128, 256, 0, stream>>>(xc, xc, 64, 64, M, partial);
  bn_finalize_kernel<<<1, 64, 0, stream>>>(partial, 128, g3, b3, 64, 1.0f / M, ss + 2 * 128);
  bn_relu_cvt_kernel<<<cvt_blocks((size_t)(M + 1) * 64), 256, 0, stream>>>(
      xc, xc, 64, 64, M, ss + 2 * 128, fbuf);
  launch_conv<64, 64>(fbuf, nbr_coarse, wp2, xc, M, M, 27, stream);

  // ---- stage 4: BN+ReLU(xc) -> transpose conv (as 8-tap gather) 64->32 -> xf ----
  bn_reduce_kernel<<<128, 256, 0, stream>>>(xc, xc, 64, 64, M, partial);
  bn_finalize_kernel<<<1, 64, 0, stream>>>(partial, 128, g4, b4, 64, 1.0f / M, ss + 3 * 128);
  bn_relu_cvt_kernel<<<cvt_blocks((size_t)(M + 1) * 64), 256, 0, stream>>>(
      xc, xc, 64, 64, M, ss + 3 * 128, fbuf);
  launch_conv<64, 32>(fbuf, didx, wpu, xf, N, M, 8, stream);

  // ---- stage 5: BN+ReLU(concat(x1,xf)) -> 27-tap sub conv 64->32 -> d_out ----
  bn_reduce_kernel<<<128, 256, 0, stream>>>(x1, xf, 32, 64, N, partial);
  bn_finalize_kernel<<<1, 64, 0, stream>>>(partial, 128, g5, b5, 64, 1.0f / N, ss + 4 * 128);
  bn_relu_cvt_kernel<<<cvt_blocks((size_t)(N + 1) * 64), 256, 0, stream>>>(
      x1, xf, 32, 64, N, ss + 4 * 128, fbuf);
  launch_conv<64, 32>(fbuf, nbr_fine, wp3, (float*)d_out, N, N, 27, stream);
}
